// CrossModalAttention_14929306321750
// MI455X (gfx1250) — matrix operand order repaired and checked by reading, not compiled, not run on hardware
//
#include <hip/hip_runtime.h>
#include <hip/hip_bf16.h>
#include <math.h>

// ---------------- CDNA5 WMMA plumbing (gfx1250, wave32) ----------------
typedef _Float16 half_t;
typedef __attribute__((ext_vector_type(16))) _Float16 v16h;
typedef __attribute__((ext_vector_type(8)))  _Float16 v8h;
typedef __attribute__((ext_vector_type(8)))  float    v8f;

#define EMBED 512
#define HEADS 8
#define HD    64
#define SEQL  512
#define BTDIM 32            // B*T
#define MTOT  (BTDIM*SEQL)  // 16384

__device__ __forceinline__ v8f wmma_f16(v16h a, v16h b, v8f c) {
  // D = A(16x32) * B(32x16) + C, f32 accumulate
  return __builtin_amdgcn_wmma_f32_16x16x32_f16(false, a, false, b,
                                                (short)0, c, false, false);
}

// CDNA5 async global->LDS copy (16B per lane), tracked by ASYNCcnt.
// lds_off = LDS byte offset (low 32 bits of a flat shared-aperture address).
__device__ __forceinline__ void async_g2lds_b128(unsigned lds_off, const void* g) {
  asm volatile("global_load_async_to_lds_b128 %0, %1, off"
               :: "v"(lds_off), "v"(g) : "memory");
}
__device__ __forceinline__ void wait_asynccnt0() {
  asm volatile("s_wait_asynccnt 0x0" ::: "memory");
}
__device__ __forceinline__ unsigned lds_offset(const void* shared_ptr) {
  return (unsigned)(unsigned long long)shared_ptr;  // LDS aperture: low 32 bits = offset
}

// A-fragment (16x32 f16, MxK). Lane m = lane&15, K-base = 8*(lane>>4).
// Halves 0..7 -> K = kb..kb+7 ; halves 8..15 -> K = kb+16..kb+23.
__device__ __forceinline__ v16h frag_a_f16(const half_t* base, int stride, int lane) {
  const int m  = lane & 15;
  const int kb = (lane >> 4) << 3;
  const half_t* p = base + m * stride + kb;
  v8h lo = *(const v8h*)(p);
  v8h hi = *(const v8h*)(p + 16);
  return __builtin_shufflevector(lo, hi, 0,1,2,3,4,5,6,7,8,9,10,11,12,13,14,15);
}

// Same A-fragment but sourced from an f32 buffer (convert on load).
__device__ __forceinline__ v16h frag_a_f32(const float* base, int stride, int lane) {
  const int m  = lane & 15;
  const int kb = (lane >> 4) << 3;
  const float* p = base + m * stride + kb;
  v16h a;
#pragma unroll
  for (int h = 0; h < 8; ++h) a[h]     = (half_t)p[h];
#pragma unroll
  for (int h = 0; h < 8; ++h) a[h + 8] = (half_t)p[16 + h];
  return a;
}

// B-fragment (32x16 f16, KxN) from K-contiguous (i.e. B^T row-major) memory.
// Lane n = lane&15, K-base = 8*(lane>>4); halves 0..7 -> K = kb..kb+7, 8..15 -> K = kb+16..kb+23.
__device__ __forceinline__ v16h frag_b_f16(const half_t* base, int stride, int lane) {
  const int n  = lane & 15;
  const int kb = (lane >> 4) << 3;
  const half_t* p = base + n * stride + kb;
  v8h lo = *(const v8h*)(p);
  v8h hi = *(const v8h*)(p + 16);
  return __builtin_shufflevector(lo, hi, 0,1,2,3,4,5,6,7,8,9,10,11,12,13,14,15);
}

// ---------------- Kernel 1: weights fp32 -> f16, transposed (Wt[n][k]) ----
__global__ __launch_bounds__(256)
void wconv_kernel(const float* __restrict__ Wq, const float* __restrict__ Wk,
                  const float* __restrict__ Wv, const float* __restrict__ Wo,
                  half_t* __restrict__ Wt) {
  int idx = blockIdx.x * blockDim.x + threadIdx.x;   // over 4*512*512
  int w = idx >> 18;
  int r = idx & ((1 << 18) - 1);
  int n = r >> 9, k = r & 511;
  const float* W = (w == 0) ? Wq : (w == 1) ? Wk : (w == 2) ? Wv : Wo;
  Wt[idx] = (half_t)W[k * EMBED + n];
}

// ---------------- Kernel 2: fused Q/K/V projection GEMM -------------------
// z = 0: Q -> [bt][h][s][d] f16 ; z = 1: K same ; z = 2: V -> [bt][h][d][s]
__global__ __launch_bounds__(256)
void proj_gemm_kernel(const float* __restrict__ Xq, const float* __restrict__ Xk,
                      const float* __restrict__ Xv, const half_t* __restrict__ WtAll,
                      const float* __restrict__ bq, const float* __restrict__ bk,
                      const float* __restrict__ bv,
                      half_t* __restrict__ Qh, half_t* __restrict__ Kh,
                      half_t* __restrict__ Vt) {
  const int z = blockIdx.z;
  const float*  X    = (z == 0) ? Xq : (z == 1) ? Xk : Xv;
  const half_t* W    = WtAll + (size_t)z * EMBED * EMBED;
  const float*  bias = (z == 0) ? bq : (z == 1) ? bk : bv;
  half_t*       Out  = (z == 0) ? Qh : (z == 1) ? Kh : Vt;

  __shared__ half_t Xs[64 * 32];

  const int tid  = threadIdx.x;
  const int lane = tid & 31;
  const int wave = tid >> 5;
  const int m0 = blockIdx.x * 64;
  const int n0 = blockIdx.y * 128 + wave * 16;

  v8f acc[4];
#pragma unroll
  for (int i = 0; i < 4; ++i) acc[i] = {};

  for (int kblk = 0; kblk < EMBED; kblk += 32) {
    __syncthreads();
    // Stage 64x32 fp32 tile of X into LDS as f16 (f32->f16 conversion in
    // flight, so this path stays on the VGPR route).
#pragma unroll
    for (int i = 0; i < 2; ++i) {
      int idx4 = tid + i * 256;
      int row  = idx4 >> 3;
      int c4   = (idx4 & 7) * 4;
      float4 v = *(const float4*)(X + (size_t)(m0 + row) * EMBED + kblk + c4);
      half_t* dst = Xs + row * 32 + c4;
      dst[0] = (half_t)v.x; dst[1] = (half_t)v.y;
      dst[2] = (half_t)v.z; dst[3] = (half_t)v.w;
    }
    __syncthreads();

    v16h b = frag_b_f16(W + (size_t)n0 * EMBED + kblk, EMBED, lane);
#pragma unroll
    for (int mt = 0; mt < 4; ++mt) {
      v16h a = frag_a_f16(Xs + mt * 16 * 32, 32, lane);
      acc[mt] = wmma_f16(a, b, acc[mt]);
    }
  }

  const int n  = n0 + (lane & 15);
  const float bn = bias[n];
  const int h = n >> 6, d = n & 63;
#pragma unroll
  for (int mt = 0; mt < 4; ++mt) {
#pragma unroll
    for (int v = 0; v < 8; ++v) {
      int m  = m0 + mt * 16 + v + ((lane >> 4) << 3);
      int bt = m >> 9, s = m & 511;
      float val = acc[mt][v] + bn;
      size_t addr = (z == 2)
          ? (((size_t)(bt * HEADS + h) * HD + d) * SEQL + s)      // V transposed
          : (((size_t)(bt * HEADS + h) * SEQL + s) * HD + d);     // Q / K
      Out[addr] = (half_t)val;
    }
  }
}

// ---------------- Kernel 3: attention per (bt, head, 64-query block) ------
__global__ __launch_bounds__(256)
void attn_kernel(const half_t* __restrict__ Qh, const half_t* __restrict__ Kh,
                 const half_t* __restrict__ Vt, const float* __restrict__ modw,
                 const float* __restrict__ tsync, const int* __restrict__ qmod,
                 const int* __restrict__ kmod, half_t* __restrict__ Att) {
  const int qb = blockIdx.x;   // query block (64 rows), 0..7
  const int h  = blockIdx.y;   // head
  const int z  = blockIdx.z;   // bt index
  const int t  = z & 15;

  const int tid  = threadIdx.x;
  const int lane = tid & 31;
  const int wave = tid >> 5;

  __shared__ float  Sc[64 * 520];      // f32 scores slab (padded stride)
  __shared__ half_t Qs[64 * 64];
  __shared__ float  partial[256];
  __shared__ float  rowmax[64];
  __shared__ float  rowinv[64];

  const float a_scale = modw[(*qmod) * HEADS + h] * modw[(*kmod) * HEADS + h] * 0.125f;
  const float t_bias  = tsync[t * HEADS + h];

  const size_t headQ = ((size_t)(z * HEADS + h) * SEQL + qb * 64) * HD;
  const size_t headK = ((size_t)(z * HEADS + h) * SEQL) * HD;
  const size_t headV = ((size_t)(z * HEADS + h) * HD) * SEQL;

  // Async-stage 64x64 f16 Q block into LDS: 1024 x 16B chunks via the
  // CDNA5 direct global->LDS path (ASYNCcnt), no VGPR round-trip.
  {
    const char* src = (const char*)(Qh + headQ);
    const unsigned dst = lds_offset(Qs);
#pragma unroll
    for (int i = 0; i < 4; ++i) {
      int c = tid + i * 256;             // 0..1023
      async_g2lds_b128(dst + c * 16, src + (size_t)c * 16);
    }
    wait_asynccnt0();
  }
  __syncthreads();

  // ---- scores = (Q K^T / sqrt(hd)) * mod_w + sync  -> LDS f32 ----
  {
    const int mt = wave >> 1;                 // 4 row tiles, 2 waves each
    const half_t* Qb = Qs + mt * 16 * 64;
    for (int j = 0; j < 16; ++j) {
      int nt = (wave & 1) + 2 * j;            // 32 col tiles split over wave pair
      v8f acc = {};
#pragma unroll
      for (int kk = 0; kk < 2; ++kk) {        // hd=64 -> 2 WMMA steps
        v16h a = frag_a_f16(Qb + kk * 32, HD, lane);
        v16h b = frag_b_f16(Kh + headK + (size_t)(nt * 16) * HD + kk * 32, HD, lane);
        acc = wmma_f16(a, b, acc);
      }
      int n = nt * 16 + (lane & 15);
#pragma unroll
      for (int v = 0; v < 8; ++v) {
        int m = mt * 16 + v + ((lane >> 4) << 3);
        Sc[m * 520 + n] = acc[v] * a_scale + t_bias;
      }
    }
  }
  __syncthreads();

  // ---- spike softmax: row max ----
  {
    int r = tid >> 2, part = tid & 3;
    const float* row = Sc + r * 520 + part * 128;
    float mx = -1e30f;
    for (int c = 0; c < 128; ++c) mx = fmaxf(mx, row[c]);
    partial[tid] = mx;
  }
  __syncthreads();
  if (tid < 64)
    rowmax[tid] = fmaxf(fmaxf(partial[tid*4], partial[tid*4+1]),
                        fmaxf(partial[tid*4+2], partial[tid*4+3]));
  __syncthreads();

  // ---- sigmoid((s-m)*5) in place + row sum ----
  {
    int r = tid >> 2, part = tid & 3;
    float mx = rowmax[r];
    float* row = Sc + r * 520 + part * 128;
    float sum = 0.f;
    for (int c = 0; c < 128; ++c) {
      float p = 1.0f / (1.0f + __expf(-5.0f * (row[c] - mx)));
      row[c] = p;
      sum += p;
    }
    partial[tid] = sum;
  }
  __syncthreads();
  if (tid < 64) {
    float s = partial[tid*4] + partial[tid*4+1] + partial[tid*4+2] + partial[tid*4+3];
    rowinv[tid] = 1.0f / (s + 1e-8f);
  }
  __syncthreads();

  // ---- attended = (attn / rowsum) @ V  -> f16 [m][embed] ----
  {
    const int mt = wave >> 1;
    const int ntb = (wave & 1) * 2;           // 4 d-tiles, 2 per wave
    v8f acc0 = {}, acc1 = {};
    for (int kblk = 0; kblk < SEQL; kblk += 32) {
      v16h a  = frag_a_f32(Sc + mt * 16 * 520 + kblk, 520, lane);
      v16h b0 = frag_b_f16(Vt + headV + (size_t)(ntb * 16) * SEQL + kblk, SEQL, lane);
      v16h b1 = frag_b_f16(Vt + headV + (size_t)((ntb + 1) * 16) * SEQL + kblk, SEQL, lane);
      acc0 = wmma_f16(a, b0, acc0);
      acc1 = wmma_f16(a, b1, acc1);
    }
    int col0 = h * HD + ntb * 16 + (lane & 15);
#pragma unroll
    for (int v = 0; v < 8; ++v) {
      int ml = mt * 16 + v + ((lane >> 4) << 3);
      size_t mg = (size_t)z * SEQL + qb * 64 + ml;
      float inv = rowinv[ml];
      Att[mg * EMBED + col0]      = (half_t)(acc0[v] * inv);
      Att[mg * EMBED + col0 + 16] = (half_t)(acc1[v] * inv);
    }
  }
}

// ---------------- Kernel 4: output projection -> fp32 d_out ---------------
__global__ __launch_bounds__(256)
void out_gemm_kernel(const half_t* __restrict__ Att, const half_t* __restrict__ WoT,
                     const float* __restrict__ bo, float* __restrict__ Out) {
  __shared__ half_t As[64 * 32];
  const int tid  = threadIdx.x;
  const int lane = tid & 31;
  const int wave = tid >> 5;
  const int m0 = blockIdx.x * 64;
  const int n0 = blockIdx.y * 128 + wave * 16;
  const unsigned as_base = lds_offset(As);

  v8f acc[4];
#pragma unroll
  for (int i = 0; i < 4; ++i) acc[i] = {};

  for (int kblk = 0; kblk < EMBED; kblk += 32) {
    __syncthreads();
    {   // 64x32 f16 tile = 256 x 16B chunks, one async g->LDS copy per thread
      int row = tid >> 2;
      int c8  = (tid & 3) * 8;
      async_g2lds_b128(as_base + tid * 16,
                       Att + (size_t)(m0 + row) * EMBED + kblk + c8);
      wait_asynccnt0();
    }
    __syncthreads();

    v16h b = frag_b_f16(WoT + (size_t)n0 * EMBED + kblk, EMBED, lane);
#pragma unroll
    for (int mt = 0; mt < 4; ++mt) {
      v16h a = frag_a_f16(As + mt * 16 * 32, 32, lane);
      acc[mt] = wmma_f16(a, b, acc[mt]);
    }
  }

  const int n = n0 + (lane & 15);
  const float bn = bo[n];
#pragma unroll
  for (int mt = 0; mt < 4; ++mt) {
#pragma unroll
    for (int v = 0; v < 8; ++v) {
      int m = m0 + mt * 16 + v + ((lane >> 4) << 3);
      Out[(size_t)m * EMBED + n] = acc[mt][v] + bn;
    }
  }
}

// ---------------- host-side launcher --------------------------------------
extern "C" void kernel_launch(void* const* d_in, const int* in_sizes, int n_in,
                              void* d_out, int out_size, void* d_ws, size_t ws_size,
                              hipStream_t stream) {
  const float* Xq  = (const float*)d_in[0];
  const float* Xk  = (const float*)d_in[1];
  const float* Xv  = (const float*)d_in[2];
  const float* Wq  = (const float*)d_in[3];
  const float* bq  = (const float*)d_in[4];
  const float* Wk  = (const float*)d_in[5];
  const float* bk  = (const float*)d_in[6];
  const float* Wv  = (const float*)d_in[7];
  const float* bv  = (const float*)d_in[8];
  const float* Wo  = (const float*)d_in[9];
  const float* bo  = (const float*)d_in[10];
  const float* mw  = (const float*)d_in[11];
  const float* ts  = (const float*)d_in[12];
  const int*   qm  = (const int*)d_in[13];
  const int*   km  = (const int*)d_in[14];
  float* out = (float*)d_out;

  // workspace layout (f16 intermediates): Qh | Kh | Vt | Att | Wt(x4)
  char* ws = (char*)d_ws;
  half_t* Qh  = (half_t*)(ws + 0);
  half_t* Kh  = (half_t*)(ws + 16777216);
  half_t* Vt  = (half_t*)(ws + 33554432);
  half_t* Att = (half_t*)(ws + 50331648);
  half_t* Wt  = (half_t*)(ws + 67108864);   // 4 * 512*512 f16 (q,k,v,o)

  wconv_kernel<<<4096, 256, 0, stream>>>(Wq, Wk, Wv, Wo, Wt);

  proj_gemm_kernel<<<dim3(MTOT / 64, EMBED / 128, 3), 256, 0, stream>>>(
      Xq, Xk, Xv, Wt, bq, bk, bv, Qh, Kh, Vt);

  attn_kernel<<<dim3(SEQL / 64, HEADS, BTDIM), 256, 0, stream>>>(
      Qh, Kh, Vt, mw, ts, qm, km, Att);

  out_gemm_kernel<<<dim3(MTOT / 64, EMBED / 128, 1), 256, 0, stream>>>(
      Att, Wt + (size_t)3 * EMBED * EMBED, bo, out);
}
